// SingleHeadAttention_50483045597977
// MI455X (gfx1250) — compile-verified, matrix operand order checked by reading, not run on hardware
//
#include <hip/hip_runtime.h>
#include <hip/hip_bf16.h>
#include <stdint.h>

// ---------------------------------------------------------------------------
// Causal single-head attention for MI455X (gfx1250, wave32, WMMA).
// B=4, S=4096, D=768.
//   cvt_x: x f32 -> bf16 row-major
//   cvt_w: W f32 [k][n] -> WT bf16 [n][k]
//   qkv_gemm: async-to-LDS double-buffered weight staging + WMMA bf16;
//             32x64 C slab per wave; Q,K bf16 [s][d]; V transposed [d][s]
//   attn: flash attention, all-wave online softmax, WMMA bf16, f32 accum.
// Workspace: ~100 MB.
// ---------------------------------------------------------------------------

typedef __attribute__((ext_vector_type(16))) __bf16 v16bf;
typedef __attribute__((ext_vector_type(8)))  float  v8f;

#define WMMA_BF16(a, b, c) \
  __builtin_amdgcn_wmma_f32_16x16x32_bf16(false, (a), false, (b), (short)0, (c), false, false)

static constexpr int Bn = 4;
static constexpr int Sn = 4096;
static constexpr int Dn = 768;
static constexpr int Mtot = Bn * Sn;   // 16384

union AB {
  uint4        q[2];   // two b128 loads
  unsigned int u[8];
  v16bf        v;      // 16 bf16 = 8 VGPRs
};

__device__ __forceinline__ unsigned short f2bf(float f) {
  unsigned int u = __float_as_uint(f);
  u = (u + 0x7FFFu + ((u >> 16) & 1u)) >> 16;   // round-to-nearest-even
  return (unsigned short)u;
}

// Per-lane async copy: LDS[lds_off .. +16) = MEM[gaddr .. +16). ASYNCcnt++.
__device__ __forceinline__ void async_load_b128(unsigned lds_off, const void* gaddr) {
  asm volatile("global_load_async_to_lds_b128 %0, %1, off"
               :: "v"(lds_off), "v"(gaddr) : "memory");
}
__device__ __forceinline__ void wait_async_le2() {
  asm volatile("s_wait_asynccnt 0x2" ::: "memory");
}
__device__ __forceinline__ void wait_async_le0() {
  asm volatile("s_wait_asynccnt 0x0" ::: "memory");
}

// ---------------------------------------------------------------- cvt_x ----
__global__ void cvt_x_kernel(const float* __restrict__ x,
                             unsigned short* __restrict__ xb) {
  int i = (blockIdx.x * 256 + threadIdx.x) * 4;
  float4 v = *(const float4*)(x + i);
  unsigned int p0 = (unsigned int)f2bf(v.x) | ((unsigned int)f2bf(v.y) << 16);
  unsigned int p1 = (unsigned int)f2bf(v.z) | ((unsigned int)f2bf(v.w) << 16);
  *(uint2*)(xb + i) = make_uint2(p0, p1);
}

// ---------------------------------------------------------------- cvt_w ----
// WT[z][n][k] = W_z[k][n]  (bf16)
__global__ void cvt_w_kernel(const float* __restrict__ Wq,
                             const float* __restrict__ Wk,
                             const float* __restrict__ Wv,
                             unsigned short* __restrict__ WT) {
  int idx = blockIdx.x * 256 + threadIdx.x;        // 0 .. D*D-1
  int z = blockIdx.z;
  const float* W = (z == 0) ? Wq : ((z == 1) ? Wk : Wv);
  int n = idx / Dn, k = idx % Dn;
  WT[(size_t)z * Dn * Dn + idx] = f2bf(W[(size_t)k * Dn + n]);
}

// ------------------------------------------------------------- qkv_gemm ----
// 8 waves; wave w computes rows [by*256 + w*32, +32) x cols [bx*64, +64).
// Weight k-slabs (64 n x 64 k bf16 = 8 KB) staged into LDS with
// global_load_async_to_lds_b128, double-buffered on ASYNCcnt.
__global__ __launch_bounds__(256) void qkv_gemm_kernel(
    const unsigned short* __restrict__ xb,   // [Mtot][Dn] bf16
    const unsigned short* __restrict__ WT,   // [3][Dn][Dn] bf16 (n-major)
    unsigned short* __restrict__ Qb,         // [Mtot][Dn]
    unsigned short* __restrict__ Kb,         // [Mtot][Dn]
    unsigned short* __restrict__ VTb) {      // [Bn][Dn][Sn]
  __shared__ unsigned short Bsm[2][64 * 64]; // 2 x 8 KB weight slabs

  const int z    = blockIdx.z;
  const int t    = threadIdx.x;
  const int wave = t >> 5;
  const int lane = t & 31;
  const int lo   = lane & 15;
  const int hi   = lane >> 4;            // 0 or 1

  const int rbase = blockIdx.y * 256 + wave * 32;
  const int nbase = blockIdx.x * 64;

  const unsigned short* Wz = WT + (size_t)z * Dn * Dn;
  const unsigned short* arow0 = xb + (size_t)(rbase + lo) * Dn;
  const unsigned short* arow1 = xb + (size_t)(rbase + 16 + lo) * Dn;

  // staging map: 512 b128 segments per slab; thread t does segs t and t+256.
  // seg s: n = s>>3, k-off = (s&7)*8 elements.
  const int sA_n = t >> 3,          sA_k = (t & 7) * 8;
  const int sB_n = (t + 256) >> 3,  sB_k = ((t + 256) & 7) * 8;

  v8f acc[8];                            // [mt][nt] = acc[mt*4+nt]
#pragma unroll
  for (int q = 0; q < 8; ++q)
#pragma unroll
    for (int r = 0; r < 8; ++r) acc[q][r] = 0.0f;

  // prime stage 0
  {
    const unsigned short* gA = Wz + (size_t)(nbase + sA_n) * Dn + 0 + sA_k;
    const unsigned short* gB = Wz + (size_t)(nbase + sB_n) * Dn + 0 + sB_k;
    unsigned lA = (unsigned)(uintptr_t)&Bsm[0][sA_n * 64 + sA_k];
    unsigned lB = (unsigned)(uintptr_t)&Bsm[0][sB_n * 64 + sB_k];
    async_load_b128(lA, gA);
    async_load_b128(lB, gB);
  }

  int buf = 0;
  for (int kc = 0; kc < Dn; kc += 64) {
    const bool more = (kc + 64) < Dn;
    if (more) {
      const unsigned short* gA = Wz + (size_t)(nbase + sA_n) * Dn + (kc + 64) + sA_k;
      const unsigned short* gB = Wz + (size_t)(nbase + sB_n) * Dn + (kc + 64) + sB_k;
      unsigned lA = (unsigned)(uintptr_t)&Bsm[buf ^ 1][sA_n * 64 + sA_k];
      unsigned lB = (unsigned)(uintptr_t)&Bsm[buf ^ 1][sB_n * 64 + sB_k];
      async_load_b128(lA, gA);
      async_load_b128(lB, gB);
      wait_async_le2();                 // this wave's current slab is in LDS
    } else {
      wait_async_le0();
    }
    __syncthreads();                    // whole block's slab is in LDS

#pragma unroll
    for (int kk = 0; kk < 2; ++kk) {    // two 32-k WMMA steps per slab
      const int kg = kc + kk * 32 + (hi ? 8 : 0);
      AB a[2];
      a[0].q[0] = *(const uint4*)(arow0 + kg);
      a[0].q[1] = *(const uint4*)(arow0 + kg + 16);
      a[1].q[0] = *(const uint4*)(arow1 + kg);
      a[1].q[1] = *(const uint4*)(arow1 + kg + 16);
      AB bb[4];                          // all B operands in distinct regs
#pragma unroll
      for (int nt = 0; nt < 4; ++nt) {
        const unsigned short* brow =
            &Bsm[buf][(nt * 16 + lo) * 64 + kk * 32 + (hi ? 8 : 0)];
        bb[nt].q[0] = *(const uint4*)brow;         // ds_load_b128
        bb[nt].q[1] = *(const uint4*)(brow + 16);
      }
#pragma unroll
      for (int mt = 0; mt < 2; ++mt)
#pragma unroll
        for (int nt = 0; nt < 4; ++nt)
          acc[mt * 4 + nt] = WMMA_BF16(a[mt].v, bb[nt].v, acc[mt * 4 + nt]);
    }
    __syncthreads();                    // done reading before slab is reused
    buf ^= 1;
  }

  if (z < 2) {
    unsigned short* out = (z == 0) ? Qb : Kb;
#pragma unroll
    for (int mt = 0; mt < 2; ++mt)
#pragma unroll
      for (int nt = 0; nt < 4; ++nt) {
        const int col = nbase + nt * 16 + lo;
#pragma unroll
        for (int r = 0; r < 8; ++r) {
          const int row = rbase + mt * 16 + r + hi * 8;
          out[(size_t)row * Dn + col] = f2bf(acc[mt * 4 + nt][r]);
        }
      }
  } else {
    // Transposed store: lane holds 8 consecutive s-rows of one d-column.
#pragma unroll
    for (int mt = 0; mt < 2; ++mt) {
      const int srow0 = rbase + mt * 16 + hi * 8;
      const int b  = srow0 / Sn;         // 256-row blocks never straddle batch
      const int s0 = srow0 - b * Sn;
#pragma unroll
      for (int nt = 0; nt < 4; ++nt) {
        const v8f& ac = acc[mt * 4 + nt];
        const int col = nbase + nt * 16 + lo;
        unsigned short* dst = VTb + ((size_t)b * Dn + col) * Sn + s0;
        unsigned int pk0 = (unsigned int)f2bf(ac[0]) | ((unsigned int)f2bf(ac[1]) << 16);
        unsigned int pk1 = (unsigned int)f2bf(ac[2]) | ((unsigned int)f2bf(ac[3]) << 16);
        unsigned int pk2 = (unsigned int)f2bf(ac[4]) | ((unsigned int)f2bf(ac[5]) << 16);
        unsigned int pk3 = (unsigned int)f2bf(ac[6]) | ((unsigned int)f2bf(ac[7]) << 16);
        *(uint4*)dst = make_uint4(pk0, pk1, pk2, pk3);
      }
    }
  }
}

// ----------------------------------------------------------------- attn ----
// Block: 256 threads = 8 waves; 32 query rows; key blocks of 64.
__global__ __launch_bounds__(256) void attn_kernel(
    const unsigned short* __restrict__ Qb,   // [Bn*Sn][Dn] bf16
    const unsigned short* __restrict__ Kb,   // [Bn*Sn][Dn] bf16
    const unsigned short* __restrict__ VTb,  // [Bn][Dn][Sn] bf16
    float* __restrict__ out) {               // [Bn*Sn][Dn] f32
  __shared__ float        S_buf[32 * 64];    // score tile (f32)
  __shared__ unsigned int P_buf[32 * 32];    // prob tile, 32x64 bf16 packed
  __shared__ float        alpha_s[32];
  __shared__ float        l_s[32];

  const int wave = threadIdx.x >> 5;
  const int lane = threadIdx.x & 31;
  const int lo   = lane & 15;
  const int hi   = lane >> 4;

  const int b  = blockIdx.y;
  const int q0 = blockIdx.x * 32;

  const float scale = rsqrtf((float)Dn);

  // Phase-1 score-tile assignment: wave -> (smi, sni) in 2x4 grid
  const int smi = wave >> 2;
  const int sni = wave & 3;
  // Phase-3 O ownership: all 32 rows x 96 columns per wave
  const int cbase = wave * 96;
  // Phase-2 softmax assignment: wave owns rows [wave*4, +4), 8 lanes per row
  const int srow  = wave * 4 + (lane >> 3);  // 0..31, stable across key loop
  const int cgrp  = lane & 7;
  const int c0    = cgrp * 8;

  v8f O[12];
#pragma unroll
  for (int tt = 0; tt < 12; ++tt)
#pragma unroll
    for (int r = 0; r < 8; ++r) O[tt][r] = 0.0f;

  float m_run = -__builtin_inff();
  float l_run = 0.0f;

  const unsigned short* qrow = Qb + ((size_t)b * Sn + q0 + smi * 16 + lo) * Dn;
  const int qmax = q0 + 31;

  for (int j0 = 0; j0 <= qmax; j0 += 64) {
    // ---- Phase 1: S = Q K^T (one 16x16 tile per wave, 24 WMMAs) ----
    const unsigned short* krow = Kb + ((size_t)b * Sn + j0 + sni * 16 + lo) * Dn;
    v8f sc;
#pragma unroll
    for (int r = 0; r < 8; ++r) sc[r] = 0.0f;

    for (int kc = 0; kc < Dn; kc += 32) {
      const int ka = kc + (hi ? 8 : 0);
      AB a, kk;
      a.q[0]  = *(const uint4*)(qrow + ka);
      a.q[1]  = *(const uint4*)(qrow + ka + 16);
      kk.q[0] = *(const uint4*)(krow + ka);
      kk.q[1] = *(const uint4*)(krow + ka + 16);
      sc = WMMA_BF16(a.v, kk.v, sc);
    }
    {
      const int col = sni * 16 + lo;
#pragma unroll
      for (int r = 0; r < 8; ++r) S_buf[(smi * 16 + r + hi * 8) * 64 + col] = sc[r];
    }
    __syncthreads();

    // ---- Phase 2: online softmax, all 8 waves (4 rows/wave, 8 lanes/row) --
    {
      const int q_abs = q0 + srow;

      float pmax = -__builtin_inff();
#pragma unroll
      for (int c = 0; c < 8; ++c) {
        const int key = j0 + c0 + c;
        float s = S_buf[srow * 64 + c0 + c] * scale;
        s = (key <= q_abs) ? s : -__builtin_inff();
        pmax = fmaxf(pmax, s);
      }
      pmax = fmaxf(pmax, __shfl_xor(pmax, 1, 32));
      pmax = fmaxf(pmax, __shfl_xor(pmax, 2, 32));
      pmax = fmaxf(pmax, __shfl_xor(pmax, 4, 32));

      const float m_new = fmaxf(m_run, pmax);
      const float a_fac = __expf(m_run - m_new);   // first iter: exp(-inf)=0

      float psum = 0.0f;
      unsigned int* prow = P_buf + srow * 32 + (c0 >> 1);
#pragma unroll
      for (int c = 0; c < 8; c += 2) {
        const int key0 = j0 + c0 + c;
        float s0v = S_buf[srow * 64 + c0 + c] * scale;
        float s1v = S_buf[srow * 64 + c0 + c + 1] * scale;
        float p0 = (key0     <= q_abs) ? __expf(s0v - m_new) : 0.0f;
        float p1 = (key0 + 1 <= q_abs) ? __expf(s1v - m_new) : 0.0f;
        psum += p0 + p1;
        prow[c >> 1] = (unsigned int)f2bf(p0) | ((unsigned int)f2bf(p1) << 16);
      }
      psum += __shfl_xor(psum, 1, 32);
      psum += __shfl_xor(psum, 2, 32);
      psum += __shfl_xor(psum, 4, 32);

      l_run = l_run * a_fac + psum;
      m_run = m_new;
      if (cgrp == 0) alpha_s[srow] = a_fac;
    }
    __syncthreads();

    // ---- Phase 3: O = alpha*O + P V (24 WMMAs per wave) ----
#pragma unroll
    for (int mt = 0; mt < 2; ++mt)
#pragma unroll
      for (int r = 0; r < 8; ++r) {
        const float af = alpha_s[mt * 16 + r + hi * 8];
#pragma unroll
        for (int nt = 0; nt < 6; ++nt) O[mt * 6 + nt][r] *= af;
      }

#pragma unroll
    for (int kc2 = 0; kc2 < 2; ++kc2) {           // two 32-key sub-chunks
      const int ka = kc2 * 32 + (hi ? 8 : 0);
      AB pa[2];
#pragma unroll
      for (int mt = 0; mt < 2; ++mt) {
        const unsigned short* prow16 =
            (const unsigned short*)P_buf + (mt * 16 + lo) * 64;
        pa[mt].q[0] = *(const uint4*)(prow16 + ka);        // ds_load_b128
        pa[mt].q[1] = *(const uint4*)(prow16 + ka + 16);
      }
      AB vb[6];                          // all V operands in distinct regs
#pragma unroll
      for (int nt = 0; nt < 6; ++nt) {
        const unsigned short* vrow =
            VTb + ((size_t)b * Dn + cbase + nt * 16 + lo) * Sn + j0;
        vb[nt].q[0] = *(const uint4*)(vrow + ka);
        vb[nt].q[1] = *(const uint4*)(vrow + ka + 16);
      }
#pragma unroll
      for (int nt = 0; nt < 6; ++nt)
#pragma unroll
        for (int mt = 0; mt < 2; ++mt)
          O[mt * 6 + nt] = WMMA_BF16(pa[mt].v, vb[nt].v, O[mt * 6 + nt]);
    }
  }

  if (cgrp == 0) l_s[srow] = l_run;
  __syncthreads();

  // ---- epilogue: O / l -> out (f32) ----
#pragma unroll
  for (int mt = 0; mt < 2; ++mt)
#pragma unroll
    for (int r = 0; r < 8; ++r) {
      const int row = mt * 16 + r + hi * 8;
      const float rl = 1.0f / l_s[row];
      float* orow = out + ((size_t)b * Sn + q0 + row) * Dn;
#pragma unroll
      for (int nt = 0; nt < 6; ++nt)
        orow[cbase + nt * 16 + lo] = O[mt * 6 + nt][r] * rl;
    }
}

// --------------------------------------------------------------------------
extern "C" void kernel_launch(void* const* d_in, const int* in_sizes, int n_in,
                              void* d_out, int out_size, void* d_ws, size_t ws_size,
                              hipStream_t stream) {
  const float* x  = (const float*)d_in[0];
  const float* Wq = (const float*)d_in[1];
  const float* Wk = (const float*)d_in[2];
  const float* Wv = (const float*)d_in[3];

  char* ws = (char*)d_ws;
  size_t off = 0;
  unsigned short* xb  = (unsigned short*)(ws + off); off += (size_t)Mtot * Dn * 2;
  unsigned short* WT  = (unsigned short*)(ws + off); off += (size_t)3 * Dn * Dn * 2;
  unsigned short* Qb  = (unsigned short*)(ws + off); off += (size_t)Mtot * Dn * 2;
  unsigned short* Kb  = (unsigned short*)(ws + off); off += (size_t)Mtot * Dn * 2;
  unsigned short* VTb = (unsigned short*)(ws + off); off += (size_t)Mtot * Dn * 2;
  (void)ws_size; (void)in_sizes; (void)n_in; (void)out_size;

  cvt_x_kernel<<<dim3((Mtot * Dn) / 1024), dim3(256), 0, stream>>>(x, xb);
  cvt_w_kernel<<<dim3((Dn * Dn) / 256, 1, 3), dim3(256), 0, stream>>>(Wq, Wk, Wv, WT);
  qkv_gemm_kernel<<<dim3(Dn / 64, Mtot / 256, 3), dim3(256), 0, stream>>>(
      xb, WT, Qb, Kb, VTb);
  attn_kernel<<<dim3(Sn / 32, Bn), dim3(256), 0, stream>>>(
      Qb, Kb, VTb, (float*)d_out);
}
